// RSSM_91087666413926
// MI455X (gfx1250) — matrix operand order repaired and checked by reading, compiled, not run on hardware
//
#include <hip/hip_runtime.h>

typedef __attribute__((ext_vector_type(16))) __bf16 v16bf;
typedef __attribute__((ext_vector_type(8)))  float  v8f;

constexpr int Bb = 128, Tt = 256, Aa = 16, Ll = 256, Ss = 32, Dd = 512, Hh = 512;

__device__ inline unsigned short f2bf(float f) {
  unsigned int u = __builtin_bit_cast(unsigned int, f);
  unsigned int r = u + 0x7FFFu + ((u >> 16) & 1u);   // round-to-nearest-even
  return (unsigned short)(r >> 16);
}
__device__ inline float bf2f(unsigned short s) {
  unsigned int u = ((unsigned int)s) << 16;
  return __builtin_bit_cast(float, u);
}
__device__ inline float sigm(float x) { return 1.0f / (1.0f + __expf(-x)); }

union FragBF { v16bf v; unsigned int w[8]; };

// Register-blocked MTxNT macro-tile of A[M,K](bf16,row-major) @ W[K,N], W stored
// pre-transposed as WT[N, ldw] (bf16, row-major). K multiple of 32.
// Fragment layouts per CDNA5 ISA 7.12.2 (16-bit A 16x32, 16-bit B 32x16, f32 C/D).
template <int MT, int NT>
__device__ inline void wmma_mt(const unsigned short* __restrict__ A, int lda,
                               const unsigned short* __restrict__ WT, int ldw,
                               int K, int tm0, int tn0, v8f (&acc)[MT][NT]) {
  const int lane = threadIdx.x & 31;
  const int mrow = lane & 15;
  const int half = lane >> 4;
  const unsigned short* arow[MT];
  const unsigned short* wrow[NT];
#pragma unroll
  for (int mi = 0; mi < MT; ++mi)
    arow[mi] = A + (size_t)((tm0 + mi) * 16 + mrow) * lda;
#pragma unroll
  for (int nj = 0; nj < NT; ++nj)
    wrow[nj] = WT + (size_t)((tn0 + nj) * 16 + mrow) * ldw;

  for (int kc = 0; kc < K; kc += 32) {
    FragBF a[MT], b[NT];
#pragma unroll
    for (int mi = 0; mi < MT; ++mi)
#pragma unroll
      for (int v = 0; v < 8; ++v) {
        int ka = kc + ((v < 4) ? 0 : 16) + half * 8 + 2 * (v & 3);
        a[mi].w[v] = *(const unsigned int*)(arow[mi] + ka);
      }
#pragma unroll
    for (int nj = 0; nj < NT; ++nj)
#pragma unroll
      for (int v = 0; v < 8; ++v) {
        int kb = kc + ((v < 4) ? 0 : 16) + half * 8 + 2 * (v & 3);
        b[nj].w[v] = *(const unsigned int*)(wrow[nj] + kb);
      }
#pragma unroll
    for (int mi = 0; mi < MT; ++mi)
#pragma unroll
      for (int nj = 0; nj < NT; ++nj)
        acc[mi][nj] = __builtin_amdgcn_wmma_f32_16x16x32_bf16(
            false, a[mi].v, false, b[nj].v, (short)0, acc[mi][nj], false, false);
  }
}

// ---------------- prologue kernels ----------------

// WT[n*Kpad + k] = bf16(W[k*N + n]); zero-pad k in [K, Kpad)
__global__ void k_convT(const float* __restrict__ W, unsigned short* __restrict__ WT,
                        int K, int N, int Kpad) {
  int idx = blockIdx.x * blockDim.x + threadIdx.x;
  if (idx >= N * Kpad) return;
  int n = idx / Kpad, k = idx - n * Kpad;
  WT[idx] = (k < K) ? f2bf(W[(size_t)k * N + n]) : (unsigned short)0;
}

__global__ void k_init(const float* __restrict__ h0, const float* __restrict__ z0,
                       float* cur_h, float* cur_z, unsigned short* ho) {
  int idx = blockIdx.x * blockDim.x + threadIdx.x;
  if (idx >= Bb * Dd) return;
  int m = idx >> 9, n = idx & 511;
  float v = h0[idx];
  cur_h[idx] = v;
  ho[(size_t)m * (Dd + Ll) + n] = f2bf(v);
  if (n < Ss) cur_z[m * Ss + n] = z0[m * Ss + n];
}

// ---------------- per-step kernels ----------------

// xz[128,64] = bf16([z | a | 0-pad])
__global__ void k_prep_x(const float* __restrict__ cur_z, const float* __restrict__ actions,
                         int t, unsigned short* __restrict__ xz) {
  int idx = blockIdx.x * blockDim.x + threadIdx.x;
  if (idx >= Bb * 64) return;
  int m = idx >> 6, n = idx & 63;
  float v = 0.f;
  if (n < Ss)           v = cur_z[m * Ss + n];
  else if (n < Ss + Aa) v = actions[((size_t)m * Tt + t) * Aa + (n - Ss)];
  xz[idx] = f2bf(v);
}

// gi = xz@Wih + bih ; gh = h@Whh + bhh   (both [128,1536], f32)
__global__ void k_gru_gemm(const unsigned short* __restrict__ xz,
                           const unsigned short* __restrict__ WihT,
                           const unsigned short* __restrict__ ho,
                           const unsigned short* __restrict__ WhhT,
                           const float* __restrict__ bih, const float* __restrict__ bhh,
                           float* __restrict__ gi, float* __restrict__ gh) {
  int tile = blockIdx.x * 4 + (threadIdx.x >> 5);
  if (tile >= 4 * 48) return;                       // 4x48 macro-tiles of 2x2
  int tm0 = (tile & 3) * 2, tn0 = (tile >> 2) * 2;
  v8f ai[2][2] = {}, ah[2][2] = {};
  wmma_mt<2, 2>(xz, 64, WihT, 64, 64, tm0, tn0, ai);        // K padded 48->64
  wmma_mt<2, 2>(ho, Dd + Ll, WhhT, Dd, Dd, tm0, tn0, ah);   // h lives in ho[:, :512]
  int lane = threadIdx.x & 31, n = lane & 15, half = lane >> 4;
#pragma unroll
  for (int mi = 0; mi < 2; ++mi)
#pragma unroll
    for (int nj = 0; nj < 2; ++nj) {
      int col = (tn0 + nj) * 16 + n;
      float bi = bih[col], bh = bhh[col];
#pragma unroll
      for (int v = 0; v < 8; ++v) {
        int row = (tm0 + mi) * 16 + v + 8 * half;
        gi[(size_t)row * 1536 + col] = ai[mi][nj][v] + bi;
        gh[(size_t)row * 1536 + col] = ah[mi][nj][v] + bh;
      }
    }
}

// GRU gates -> new h; fill ho = [h|o_t] (bf16) and st[:, :512] = h (bf16)
__global__ void k_gru_gate(const float* __restrict__ gi, const float* __restrict__ gh,
                           const float* __restrict__ obs_e, int t,
                           float* __restrict__ cur_h, unsigned short* __restrict__ ho,
                           unsigned short* __restrict__ st, float* __restrict__ h_seq) {
  int idx = blockIdx.x * blockDim.x + threadIdx.x;
  if (idx >= Bb * Dd) return;
  int m = idx >> 9, n = idx & 511;
  size_t r0 = (size_t)m * 1536;
  float r  = sigm(gi[r0 + n] + gh[r0 + n]);
  float u  = sigm(gi[r0 + 512 + n] + gh[r0 + 512 + n]);
  float nn = tanhf(gi[r0 + 1024 + n] + r * gh[r0 + 1024 + n]);
  float hn = (1.0f - u) * nn + u * cur_h[idx];
  cur_h[idx] = hn;
  unsigned short hb = f2bf(hn);
  ho[(size_t)m * (Dd + Ll) + n] = hb;
  st[(size_t)m * (Dd + Ss) + n] = hb;
  h_seq[((size_t)m * Tt + t) * Dd + n] = hn;
  if (n < Ll) ho[(size_t)m * (Dd + Ll) + Dd + n] =
      f2bf(obs_e[((size_t)m * Tt + t) * Ll + n]);
}

struct L1Job {
  const unsigned short* A; int lda, K;
  const unsigned short* WT; int ldw;
  const float* b; unsigned short* out;
};

// out_bf16[128,512] = relu(A@W + b), job selected by blockIdx.z
__global__ void k_layer1(L1Job j0, L1Job j1, L1Job j2) {
  L1Job j = (blockIdx.z == 0) ? j0 : ((blockIdx.z == 1) ? j1 : j2);
  int tile = blockIdx.x * 4 + (threadIdx.x >> 5);
  if (tile >= 4 * 16) return;                       // 4x16 macro-tiles of 2x2
  int tm0 = (tile & 3) * 2, tn0 = (tile >> 2) * 2;
  v8f acc[2][2] = {};
  wmma_mt<2, 2>(j.A, j.lda, j.WT, j.ldw, j.K, tm0, tn0, acc);
  int lane = threadIdx.x & 31, n = lane & 15, half = lane >> 4;
#pragma unroll
  for (int mi = 0; mi < 2; ++mi)
#pragma unroll
    for (int nj = 0; nj < 2; ++nj) {
      int col = (tn0 + nj) * 16 + n;
      float b = j.b[col];
#pragma unroll
      for (int v = 0; v < 8; ++v)
        j.out[(size_t)((tm0 + mi) * 16 + v + 8 * half) * Hh + col] =
            f2bf(fmaxf(acc[mi][nj][v] + b, 0.f));
    }
}

// second layers of posterior/prior: [128,512]@[512,64] -> raw f32 stats buffer
__global__ void k_gemm2qp(const unsigned short* qh1, const unsigned short* qW2T,
                          const float* qb2, float* qpost,
                          const unsigned short* ph1, const unsigned short* pW2T,
                          const float* pb2, float* ppri) {
  const unsigned short* A  = blockIdx.z ? ph1  : qh1;
  const unsigned short* WT = blockIdx.z ? pW2T : qW2T;
  const float* b           = blockIdx.z ? pb2  : qb2;
  float* out               = blockIdx.z ? ppri : qpost;
  int tile = blockIdx.x * 4 + (threadIdx.x >> 5);
  if (tile >= 4 * 2) return;                        // 4x2 macro-tiles of 2x2
  int tm0 = (tile & 3) * 2, tn0 = (tile >> 2) * 2;
  v8f acc[2][2] = {};
  wmma_mt<2, 2>(A, Hh, WT, Hh, Hh, tm0, tn0, acc);
  int lane = threadIdx.x & 31, n = lane & 15, half = lane >> 4;
#pragma unroll
  for (int mi = 0; mi < 2; ++mi)
#pragma unroll
    for (int nj = 0; nj < 2; ++nj) {
      int col = (tn0 + nj) * 16 + n;
      float bb = b[col];
#pragma unroll
      for (int v = 0; v < 8; ++v)
        out[(size_t)((tm0 + mi) * 16 + v + 8 * half) * 64 + col] = acc[mi][nj][v] + bb;
    }
}

// split mu/logstd, clip, sample z, write stat outputs + z into st / cur_z
__global__ void k_stats(const float* __restrict__ qpost, const float* __restrict__ ppri,
                        const float* __restrict__ eps, int t,
                        float* cur_z, unsigned short* st,
                        float* qmu_o, float* qls_o, float* pmu_o, float* pls_o,
                        float* z_o) {
  int idx = blockIdx.x * blockDim.x + threadIdx.x;
  if (idx >= Bb * Ss) return;
  int m = idx >> 5, s = idx & 31;
  size_t oo = ((size_t)m * Tt + t) * Ss + s;
  float mu = qpost[m * 64 + s];
  float ls = fminf(fmaxf(qpost[m * 64 + 32 + s], -10.f), 2.f);
  float z  = mu + eps[oo] * __expf(ls);
  qmu_o[oo] = mu;  qls_o[oo] = ls;  z_o[oo] = z;
  cur_z[m * Ss + s] = z;
  st[(size_t)m * (Dd + Ss) + Dd + s] = f2bf(z);
  float pm = ppri[m * 64 + s];
  float pl = fminf(fmaxf(ppri[m * 64 + 32 + s], -10.f), 2.f);
  pmu_o[oo] = pm;  pls_o[oo] = pl;
}

// z==0: obs head GEMM [128,512]@[512,256]; z==1/2: reward/done dot products
__global__ void k_heads2(const unsigned short* oh1, const unsigned short* oW2T,
                         const float* ob2,
                         const unsigned short* rh1, const unsigned short* rW2T,
                         const float* rb2,
                         const unsigned short* dh1, const unsigned short* dW2T,
                         const float* db2,
                         float* obs_o, float* rew_o, float* done_o, int t) {
  if (blockIdx.z == 0) {
    int tile = blockIdx.x * 4 + (threadIdx.x >> 5);
    if (tile >= 4 * 8) return;                      // 4x8 macro-tiles of 2x2
    int tm0 = (tile & 3) * 2, tn0 = (tile >> 2) * 2;
    v8f acc[2][2] = {};
    wmma_mt<2, 2>(oh1, Hh, oW2T, Hh, Hh, tm0, tn0, acc);
    int lane = threadIdx.x & 31, n = lane & 15, half = lane >> 4;
#pragma unroll
    for (int mi = 0; mi < 2; ++mi)
#pragma unroll
      for (int nj = 0; nj < 2; ++nj) {
        int col = (tn0 + nj) * 16 + n;
        float b = ob2[col];
#pragma unroll
        for (int v = 0; v < 8; ++v) {
          int m = (tm0 + mi) * 16 + v + 8 * half;
          obs_o[((size_t)m * Tt + t) * Ll + col] = acc[mi][nj][v] + b;
        }
      }
  } else {
    const unsigned short* h = (blockIdx.z == 1) ? rh1  : dh1;
    const unsigned short* w = (blockIdx.z == 1) ? rW2T : dW2T;
    const float* b          = (blockIdx.z == 1) ? rb2  : db2;
    float* out              = (blockIdx.z == 1) ? rew_o : done_o;
    int m = blockIdx.x * 4 + (threadIdx.x >> 5);   // one wave per batch row
    int lane = threadIdx.x & 31;
    const unsigned short* row = h + (size_t)m * Hh;
    float s = 0.f;
#pragma unroll
    for (int i = 0; i < 16; ++i) {
      int k = lane * 16 + i;
      s += bf2f(row[k]) * bf2f(w[k]);
    }
    for (int off = 16; off; off >>= 1) s += __shfl_xor(s, off, 32);
    if (lane == 0) out[(size_t)m * Tt + t] = s + b[0];
  }
}

// ---------------- host ----------------

extern "C" void kernel_launch(void* const* d_in, const int* in_sizes, int n_in,
                              void* d_out, int out_size, void* d_ws, size_t ws_size,
                              hipStream_t stream) {
  (void)in_sizes; (void)n_in; (void)out_size; (void)ws_size;
  const float* actions = (const float*)d_in[0];
  const float* obs_e   = (const float*)d_in[1];
  const float* eps     = (const float*)d_in[2];
  const float* h0      = (const float*)d_in[3];
  const float* z0      = (const float*)d_in[4];
  const float* Wih = (const float*)d_in[5],  *Whh = (const float*)d_in[6];
  const float* bih = (const float*)d_in[7],  *bhh = (const float*)d_in[8];
  const float* pW1 = (const float*)d_in[9],  *pb1 = (const float*)d_in[10];
  const float* pW2 = (const float*)d_in[11], *pb2 = (const float*)d_in[12];
  const float* qW1 = (const float*)d_in[13], *qb1 = (const float*)d_in[14];
  const float* qW2 = (const float*)d_in[15], *qb2 = (const float*)d_in[16];
  const float* oW1 = (const float*)d_in[17], *ob1 = (const float*)d_in[18];
  const float* oW2 = (const float*)d_in[19], *ob2 = (const float*)d_in[20];
  const float* rW1 = (const float*)d_in[21], *rb1 = (const float*)d_in[22];
  const float* rW2 = (const float*)d_in[23], *rb2 = (const float*)d_in[24];
  const float* dW1 = (const float*)d_in[25], *db1 = (const float*)d_in[26];
  const float* dW2 = (const float*)d_in[27], *db2 = (const float*)d_in[28];

  // workspace carve-up (deterministic)
  char* wp = (char*)d_ws;
  auto alloc = [&](size_t bytes) {
    char* p = wp; wp += (bytes + 255) & ~(size_t)255; return p;
  };
  unsigned short* WihT = (unsigned short*)alloc(2u * 1536 * 64);
  unsigned short* WhhT = (unsigned short*)alloc(2u * 1536 * 512);
  unsigned short* qW1T = (unsigned short*)alloc(2u * 512 * 768);
  unsigned short* qW2T = (unsigned short*)alloc(2u * 64 * 512);
  unsigned short* pW1T = (unsigned short*)alloc(2u * 512 * 512);
  unsigned short* pW2T = (unsigned short*)alloc(2u * 64 * 512);
  unsigned short* oW1T = (unsigned short*)alloc(2u * 512 * 544);
  unsigned short* oW2T = (unsigned short*)alloc(2u * 256 * 512);
  unsigned short* rW1T = (unsigned short*)alloc(2u * 512 * 544);
  unsigned short* rW2T = (unsigned short*)alloc(2u * 512);
  unsigned short* dW1T = (unsigned short*)alloc(2u * 512 * 544);
  unsigned short* dW2T = (unsigned short*)alloc(2u * 512);
  unsigned short* xz   = (unsigned short*)alloc(2u * Bb * 64);
  unsigned short* ho   = (unsigned short*)alloc(2u * Bb * (Dd + Ll));
  unsigned short* st   = (unsigned short*)alloc(2u * Bb * (Dd + Ss));
  unsigned short* qh1  = (unsigned short*)alloc(2u * Bb * Hh);
  unsigned short* ph1  = (unsigned short*)alloc(2u * Bb * Hh);
  unsigned short* oh1  = (unsigned short*)alloc(2u * Bb * Hh);
  unsigned short* rh1  = (unsigned short*)alloc(2u * Bb * Hh);
  unsigned short* dh1  = (unsigned short*)alloc(2u * Bb * Hh);
  float* gi    = (float*)alloc(4u * Bb * 1536);
  float* gh    = (float*)alloc(4u * Bb * 1536);
  float* cur_h = (float*)alloc(4u * Bb * Dd);
  float* cur_z = (float*)alloc(4u * Bb * Ss);
  float* qpost = (float*)alloc(4u * Bb * 64);
  float* ppri  = (float*)alloc(4u * Bb * 64);

  // output slices (tuple order, each [B, T, ...] flat)
  float* out   = (float*)d_out;
  float* h_seq = out;
  float* z_seq = h_seq + (size_t)Bb * Tt * Dd;
  float* obs_o = z_seq + (size_t)Bb * Tt * Ss;
  float* rew_o = obs_o + (size_t)Bb * Tt * Ll;
  float* don_o = rew_o + (size_t)Bb * Tt;
  float* pmu_o = don_o + (size_t)Bb * Tt;
  float* pls_o = pmu_o + (size_t)Bb * Tt * Ss;
  float* qmu_o = pls_o + (size_t)Bb * Tt * Ss;
  float* qls_o = qmu_o + (size_t)Bb * Tt * Ss;

  auto convT = [&](const float* W, unsigned short* WT, int K, int N, int Kpad) {
    int tot = N * Kpad;
    k_convT<<<(tot + 255) / 256, 256, 0, stream>>>(W, WT, K, N, Kpad);
  };
  convT(Wih, WihT, 48, 1536, 64);
  convT(Whh, WhhT, 512, 1536, 512);
  convT(qW1, qW1T, 768, 512, 768);
  convT(qW2, qW2T, 512, 64, 512);
  convT(pW1, pW1T, 512, 512, 512);
  convT(pW2, pW2T, 512, 64, 512);
  convT(oW1, oW1T, 544, 512, 544);
  convT(oW2, oW2T, 512, 256, 512);
  convT(rW1, rW1T, 544, 512, 544);
  convT(rW2, rW2T, 512, 1, 512);
  convT(dW1, dW1T, 544, 512, 544);
  convT(dW2, dW2T, 512, 1, 512);

  k_init<<<(Bb * Dd + 255) / 256, 256, 0, stream>>>(h0, z0, cur_h, cur_z, ho);

  for (int t = 0; t < Tt; ++t) {
    k_prep_x<<<(Bb * 64 + 255) / 256, 256, 0, stream>>>(cur_z, actions, t, xz);
    k_gru_gemm<<<48, 128, 0, stream>>>(xz, WihT, ho, WhhT, bih, bhh, gi, gh);
    k_gru_gate<<<(Bb * Dd + 255) / 256, 256, 0, stream>>>(gi, gh, obs_e, t,
                                                          cur_h, ho, st, h_seq);
    L1Job jq{ho, Dd + Ll, Dd + Ll, qW1T, Dd + Ll, qb1, qh1};
    L1Job jp{ho, Dd + Ll, Dd,      pW1T, Dd,      pb1, ph1};
    k_layer1<<<dim3(16, 1, 2), 128, 0, stream>>>(jq, jp, jp);
    k_gemm2qp<<<dim3(2, 1, 2), 128, 0, stream>>>(qh1, qW2T, qb2, qpost,
                                                 ph1, pW2T, pb2, ppri);
    k_stats<<<(Bb * Ss + 255) / 256, 256, 0, stream>>>(qpost, ppri, eps, t, cur_z, st,
                                                       qmu_o, qls_o, pmu_o, pls_o, z_seq);
    L1Job jo{st, Dd + Ss, Dd + Ss, oW1T, Dd + Ss, ob1, oh1};
    L1Job jr{st, Dd + Ss, Dd + Ss, rW1T, Dd + Ss, rb1, rh1};
    L1Job jd{st, Dd + Ss, Dd + Ss, dW1T, Dd + Ss, db1, dh1};
    k_layer1<<<dim3(16, 1, 3), 128, 0, stream>>>(jo, jr, jd);
    k_heads2<<<dim3(32, 1, 3), 128, 0, stream>>>(oh1, oW2T, ob2, rh1, rW2T, rb2,
                                                 dh1, dW2T, db2, obs_o, rew_o, don_o, t);
  }
}
